// RCNN_77644418777652
// MI455X (gfx1250) — matrix operand order repaired and checked by reading, not compiled, not run on hardware
//
#include <hip/hip_runtime.h>
#include <hip/hip_bf16.h>

// Problem constants (from reference)
#define B_  64
#define L_  2048
#define V_  50000
#define E_  300
#define EP  320      // E padded to multiple of 32 (K-tile) for WMMA
#define H_  128
#define C_  10
#define LT  16       // L / 128 row-tiles per batch
#define KT  10       // EP / 32 k-tiles

typedef __attribute__((ext_vector_type(16))) __bf16 v16bf;
typedef __attribute__((ext_vector_type(8)))  float  v8f;

struct Frag32 { uint4 lo, hi; };   // 32 bytes == 16 bf16 == one WMMA operand per lane

__device__ __forceinline__ unsigned short f2bf(float f) {
  unsigned int u = __float_as_uint(f);
  u += 0x7FFFu + ((u >> 16) & 1u);           // round-to-nearest-even
  return (unsigned short)(u >> 16);
}

// monotone float<->uint encoding so atomicMin/Max on uint give float min/max
__device__ __forceinline__ unsigned int fenc(float f) {
  unsigned int u = __float_as_uint(f);
  return (u & 0x80000000u) ? ~u : (u | 0x80000000u);
}
__device__ __forceinline__ float fdec(unsigned int u) {
  unsigned int b = (u & 0x80000000u) ? (u & 0x7FFFFFFFu) : ~u;
  return __uint_as_float(b);
}

// ---------------------------------------------------------------- init ----
__global__ void k_init(unsigned int* s) {
  s[0] = 0xFFFFFFFFu;   // encoded +inf -> min accumulator
  s[1] = 0x00000000u;   // encoded -inf -> max accumulator
}

// ------------------------------------------------- global min/max of emb[x]
__global__ void __launch_bounds__(256) k_minmax(const int* __restrict__ x,
                                                const float* __restrict__ emb,
                                                unsigned int* s) {
  int lane = threadIdx.x & 31;
  int gw   = (int)((blockIdx.x * blockDim.x + threadIdx.x) >> 5);
  int nw   = (int)((gridDim.x * blockDim.x) >> 5);
  float mn = 3.4e38f, mx = -3.4e38f;
  for (int item = gw; item < B_ * L_; item += nw) {
    const float* row = emb + (size_t)x[item] * E_;
    for (int e = lane; e < E_; e += 32) {     // coalesced across the wave
      float v = row[e];
      mn = fminf(mn, v);
      mx = fmaxf(mx, v);
    }
  }
  for (int o = 16; o > 0; o >>= 1) {
    mn = fminf(mn, __shfl_xor(mn, o, 32));
    mx = fmaxf(mx, __shfl_xor(mx, o, 32));
  }
  if (lane == 0) {
    atomicMin(&s[0], fenc(mn));
    atomicMax(&s[1], fenc(mx));
  }
}

// ----------------------------------- pack W1 into WMMA dense-B bf16 layout
// B 32x16 bf16, k order used here: lanes 0-15 hold col N=lane, K=0..15; lanes 16-31 hold K=16..31.
// W1p[kt*4096 + h*32 + k] = bf16(W1[(kt*32+k)*H + h]), zero-padded past E.
__global__ void __launch_bounds__(256) k_packW1(const float* __restrict__ W1,
                                                unsigned short* __restrict__ W1p) {
  int t = blockIdx.x * blockDim.x + threadIdx.x;
  if (t >= KT * H_ * 32) return;
  int kt = t / (H_ * 32);
  int r  = t % (H_ * 32);
  int h  = r / 32;
  int k  = r % 32;
  int e  = kt * 32 + k;
  float v = (e < E_) ? W1[e * H_ + h] : 0.0f;
  W1p[t] = f2bf(v);                 // t == kt*4096 + h*32 + k exactly
}

// ------------------------------------------- recurrence -> Wm (bf16, padded)
// One block per batch b; thread e owns one chain over L.  x-row staged in LDS.
__global__ void __launch_bounds__(EP) k_recur(const int* __restrict__ x,
                                              const float* __restrict__ emb,
                                              const float* __restrict__ alpha_p,
                                              const float* __restrict__ beta_p,
                                              const unsigned int* __restrict__ s,
                                              unsigned short* __restrict__ Wm) {
  __shared__ int xrow[L_];
  int b = blockIdx.x;
  int e = threadIdx.x;
  for (int i = threadIdx.x; i < L_; i += blockDim.x) xrow[i] = x[b * L_ + i];
  __syncthreads();

  unsigned short* wrow = Wm + (size_t)b * L_ * EP + e;
  if (e >= E_) {                               // zero the K-padding lanes
    for (int l = 0; l < L_; ++l) wrow[(size_t)l * EP] = 0;
    return;
  }
  float alpha = alpha_p[0];
  float beta  = beta_p[0];
  float mn = fdec(s[0]);
  float mx = fdec(s[1]);
  float inv = 1.0f / (mx - mn);

  float w = 1.0f;
  wrow[0] = f2bf(1.0f);
  for (int k = 1; k < L_; ++k) {
    float xe = emb[(size_t)xrow[k - 1] * E_ + e];   // coalesced across e
    float xn = (xe - mn) * inv;
    w = w + alpha / w - beta * xn;
    wrow[(size_t)k * EP] = f2bf(w);
  }
}

// ------------------- WMMA GEMM [128 rows x 128 cols] + tanh + row-max fuse
// Block = (b, ltile): 8 waves, wave wv owns rows [wv*16, wv*16+16), all 8
// 16-col tiles.  A fragments stream from global (contiguous k = 16*half + s, matching B);
// B k-slices (8 KB each) are double-buffered in LDS via the CDNA5 async
// global->LDS path (global_load_async_to_lds_b128 / ASYNCcnt) and consumed
// with ds_load_b128.  The k-loop is fully unrolled so accumulators stay in
// statically assigned even-aligned VGPR tuples (no v_mov shuffles).
__global__ void __launch_bounds__(256) k_gemm(const unsigned short* __restrict__ Wm,
                                              const unsigned short* __restrict__ W1p,
                                              const float* __restrict__ b1,
                                              float* __restrict__ P) {
  __shared__ __align__(16) unsigned short Bs[2][4096];  // 2 x 8KB k-slices
  __shared__ float red[8 * H_];
  int b    = blockIdx.x / LT;
  int lt   = blockIdx.x % LT;
  int wv   = threadIdx.x >> 5;
  int lane = threadIdx.x & 31;
  int ln   = lane & 15;
  int hs   = lane >> 4;
  int t    = threadIdx.x;

  // LDS byte offset of Bs: flat shared pointers carry the aperture in
  // addr[63:32]; addr[31:0] is the LDS offset (ISA 10.2).
  unsigned lds0 = (unsigned)(size_t)&Bs[0][0];

  int row = b * L_ + lt * 128 + wv * 16 + ln;
  const unsigned short* arow = Wm + (size_t)row * EP + hs * 16;

  // Issue one 8KB k-slice prefetch: 256 threads x 2 async b128 (16B each).
  // INST_OFFSET is added to both the LDS and global addresses.
  auto prefetch = [&](int kt, int buf) {
    unsigned ldsoff = lds0 + (unsigned)(buf * 8192 + t * 16);
    int      voff   = kt * 8192 + t * 16;
    asm volatile("global_load_async_to_lds_b128 %0, %1, %2"
                 :: "v"(ldsoff), "v"(voff), "s"(W1p) : "memory");
    asm volatile("global_load_async_to_lds_b128 %0, %1, %2 offset:4096"
                 :: "v"(ldsoff), "v"(voff), "s"(W1p) : "memory");
  };

  prefetch(0, 0);

  // hoist epilogue bias loads so their latency hides under the GEMM
  float bb[8];
#pragma unroll
  for (int ct = 0; ct < 8; ++ct) bb[ct] = b1[ct * 16 + ln];

  v8f acc[8] = {};
#pragma unroll
  for (int kt = 0; kt < KT; ++kt) {
    int buf = kt & 1;
    if (kt + 1 < KT) {
      prefetch(kt + 1, buf ^ 1);                       // overlap next slice
      asm volatile("s_wait_asynccnt %0" :: "n"(2) : "memory");  // slice kt done
    } else {
      asm volatile("s_wait_asynccnt %0" :: "n"(0) : "memory");
    }
    __syncthreads();                                   // all waves' data visible

    Frag32 af;
    const unsigned short* abase = arow + kt * 32;
    af.lo = *(const uint4*)(abase);        // K chunk 0
    af.hi = *(const uint4*)(abase + 8);    // K chunk 1
    v16bf av = __builtin_bit_cast(v16bf, af);
#pragma unroll
    for (int ct = 0; ct < 8; ++ct) {
      const unsigned short* bbase = &Bs[buf][(ct * 16 + ln) * 32 + hs * 16];
      Frag32 bfg;
      bfg.lo = *(const uint4*)(bbase);     // ds_load_b128
      bfg.hi = *(const uint4*)(bbase + 8);
      v16bf bv = __builtin_bit_cast(v16bf, bfg);
      acc[ct] = __builtin_amdgcn_wmma_f32_16x16x32_bf16(
          false, av, false, bv, (short)0, acc[ct], false, false);
    }
    __syncthreads();   // protect this buffer before it is refilled next iter
  }

  // epilogue: h = tanh(acc + b1), max over this wave's 16 rows per column
  float m[8];
#pragma unroll
  for (int ct = 0; ct < 8; ++ct) {
    float mm = -3.4e38f;
#pragma unroll
    for (int i = 0; i < 8; ++i) {          // VGPR i -> rows M=i and M=i+8
      float h = tanhf(acc[ct][i] + bb[ct]);
      mm = fmaxf(mm, h);
    }
    mm = fmaxf(mm, __shfl_xor(mm, 16, 32));  // merge the two row halves
    m[ct] = mm;
  }
  if (hs == 0) {
#pragma unroll
    for (int ct = 0; ct < 8; ++ct) red[wv * H_ + ct * 16 + ln] = m[ct];
  }
  __syncthreads();
  for (int tt = threadIdx.x; tt < H_; tt += blockDim.x) {
    float p = red[tt];
#pragma unroll
    for (int w2 = 1; w2 < 8; ++w2) p = fmaxf(p, red[w2 * H_ + tt]);
    P[(size_t)blockIdx.x * H_ + tt] = p;   // per-(b,ltile) partial max
  }
}

// ------------------------- final: max over ltiles, pooled @ W2 + b2 -> out
__global__ void __launch_bounds__(128) k_final(const float* __restrict__ P,
                                               const float* __restrict__ W2,
                                               const float* __restrict__ b2,
                                               float* __restrict__ out) {
  __shared__ float pld[H_];
  int b = blockIdx.x;
  int h = threadIdx.x;
  float m = P[(size_t)(b * LT) * H_ + h];
  for (int t = 1; t < LT; ++t) m = fmaxf(m, P[(size_t)(b * LT + t) * H_ + h]);
  pld[h] = m;
  __syncthreads();
  if (h < C_) {
    float acc = b2[h];
    for (int k = 0; k < H_; ++k) acc += pld[k] * W2[k * C_ + h];
    out[b * C_ + h] = acc;
  }
}

// --------------------------------------------------------------- launcher
extern "C" void kernel_launch(void* const* d_in, const int* in_sizes, int n_in,
                              void* d_out, int out_size, void* d_ws, size_t ws_size,
                              hipStream_t stream) {
  const int*   x     = (const int*)d_in[0];
  const float* emb   = (const float*)d_in[1];
  const float* alpha = (const float*)d_in[2];
  const float* beta  = (const float*)d_in[3];
  const float* W1    = (const float*)d_in[4];
  const float* b1    = (const float*)d_in[5];
  const float* W2    = (const float*)d_in[6];
  const float* b2    = (const float*)d_in[7];
  float*       out   = (float*)d_out;

  char* ws = (char*)d_ws;
  unsigned int*   scal = (unsigned int*)(ws);                 // 2 x u32
  unsigned short* W1p  = (unsigned short*)(ws + 256);         // 81,920 B packed B-frags
  float*          P    = (float*)(ws + 256 + 81920);          // 524,288 B partials
  unsigned short* Wm   = (unsigned short*)(ws + (1 << 20));   // ~80 MB bf16 A matrix

  hipLaunchKernelGGL(k_init,   dim3(1),          dim3(1),   0, stream, scal);
  hipLaunchKernelGGL(k_minmax, dim3(2048),       dim3(256), 0, stream, x, emb, scal);
  hipLaunchKernelGGL(k_packW1, dim3((KT*H_*32 + 255)/256), dim3(256), 0, stream, W1, W1p);
  hipLaunchKernelGGL(k_recur,  dim3(B_),         dim3(EP),  0, stream,
                     x, emb, alpha, beta, scal, Wm);
  hipLaunchKernelGGL(k_gemm,   dim3(B_ * LT),    dim3(256), 0, stream, Wm, W1p, b1, P);
  hipLaunchKernelGGL(k_final,  dim3(B_),         dim3(128), 0, stream, P, W2, b2, out);
}
